// RNNCopyMechanismMixin_13022340842312
// MI455X (gfx1250) — matrix operand order repaired and checked by reading, not compiled, not run on hardware
//
#include <hip/hip_runtime.h>
#include <stdint.h>

#define B_ROWS 256
#define LAYER  1024
#define VOCAB  50257
#define SRC    400
#define BN     64     // vocab columns per block
#define BK     64     // K depth staged per pipeline stage (2 WMMA K-steps)
#define KSTAGES (LAYER / BK)   // 16

typedef __attribute__((ext_vector_type(16))) __bf16 v16bf;
typedef __attribute__((ext_vector_type(8)))  __bf16 v8bf;
typedef __attribute__((ext_vector_type(4)))  __bf16 v4bf;
typedef __attribute__((ext_vector_type(8)))  float  v8f;
typedef __attribute__((ext_vector_type(4)))  float  f4;

__device__ __forceinline__ v16bf concat8(v8bf lo, v8bf hi) {
    return __builtin_shufflevector(lo, hi, 0, 1, 2, 3, 4, 5, 6, 7,
                                   8, 9, 10, 11, 12, 13, 14, 15);
}

// ---------------------------------------------------------------------------
// K1: convert X to bf16 (WMMA A operand) and compute p_gen per row.
// ---------------------------------------------------------------------------
__global__ __launch_bounds__(256) void prep_kernel(
    const float* __restrict__ x, const float* __restrict__ Wgen,
    const float* __restrict__ bgen, uint16_t* __restrict__ Abf,
    float* __restrict__ pgen) {
    const int b = blockIdx.x;
    const int t = threadIdx.x;
    const float* row = x + (size_t)b * LAYER;
    float partial = 0.f;
#pragma unroll
    for (int i = 0; i < LAYER / 256; ++i) {
        const int k = t + i * 256;
        const float v = row[k];
        Abf[(size_t)b * LAYER + k] = __builtin_bit_cast(uint16_t, (__bf16)v);
        partial += v * Wgen[k];
    }
    __shared__ float red[256];
    red[t] = partial;
    __syncthreads();
    for (int s = 128; s > 0; s >>= 1) {
        if (t < s) red[t] += red[t + s];
        __syncthreads();
    }
    if (t == 0) {
        const float z = red[0] + bgen[0];
        pgen[b] = 1.f / (1.f + __expf(-z));
    }
}

// ---------------------------------------------------------------------------
// K2: logits = X @ W_out^T + b_out via bf16 WMMA.
// Block = 256 threads = 8 waves: 4 M-groups (64 rows) x 2 N-groups (32 cols).
// Pipeline per stage (BK=64 of K):
//   async f32 W tile -> LDS (double buffered, s_wait_asynccnt)
//   cooperative f32 -> bf16 convert into single bf16 LDS tile
//   waves: ds_load_b128 B fragments, global bf16 A fragments, 16 WMMAs.
// ---------------------------------------------------------------------------
__device__ __forceinline__ void stage_tile_async(
    const float* __restrict__ Wout, int vblock, int kbase,
    uint32_t lds_base, int t) {
    const uint64_t sbase = (uint64_t)(uintptr_t)Wout;
#pragma unroll
    for (int c2 = 0; c2 < 4; ++c2) {
        const int chunk = t + c2 * 256;      // 0..1023 (1024 x 16B = 16KB)
        const int r = chunk >> 4;            // tile row (vocab) 0..63
        const int c = chunk & 15;            // 16B chunk within K row
        int v = vblock + r;
        v = (v < VOCAB) ? v : (VOCAB - 1);   // clamp OOB rows (masked at store)
        const uint32_t voff =
            ((uint32_t)v * (uint32_t)LAYER + (uint32_t)(kbase + c * 4)) * 4u;
        const uint32_t laddr = lds_base + (uint32_t)chunk * 16u;
        asm volatile("global_load_async_to_lds_b128 %0, %1, %2"
                     :: "v"(laddr), "v"(voff), "s"(sbase)
                     : "memory");
    }
}

__global__ __launch_bounds__(256) void gemm_kernel(
    const uint16_t* __restrict__ Abf, const float* __restrict__ Wout,
    const float* __restrict__ bout, float* __restrict__ logits) {
    __shared__ float    fbuf[2][BN * BK];    // 2 x 16KB async f32 staging
    __shared__ uint16_t bbuf[BN * BK];       // 8KB bf16 tile (single buffer)

    const int t      = threadIdx.x;
    const int lane   = t & 31;
    const int wave   = t >> 5;
    const int mgroup = wave & 3;             // 4 M-groups of 64 rows
    const int ngroup = wave >> 2;            // 2 N-groups of 32 cols
    const int vblock = blockIdx.x * BN;

    const int lrow = lane & 15;
    const int sel  = (lane >> 4) & 1;        // half-wave selector

    v8f acc[4][2] = {};                      // 4 M-tiles x 2 N-tiles per wave

    stage_tile_async(Wout, vblock, 0, (uint32_t)(uintptr_t)&fbuf[0][0], t);

    for (int ks = 0; ks < KSTAGES; ++ks) {
        const int fb = ks & 1;
        if (ks + 1 < KSTAGES) {
            stage_tile_async(Wout, vblock, (ks + 1) * BK,
                             (uint32_t)(uintptr_t)&fbuf[fb ^ 1][0], t);
            // 4 just-issued loads stay in flight; current tile's 4 complete
            asm volatile("s_wait_asynccnt 4" ::: "memory");
        } else {
            asm volatile("s_wait_asynccnt 0" ::: "memory");
        }
        __syncthreads();                     // f32 tile visible to all waves

        // Cooperative f32 -> bf16: 16 contiguous elements per thread.
        {
            const int row = t >> 2;
            const int col = (t & 3) * 16;
            const float* src = &fbuf[fb][row * BK + col];
            const f4 q0 = *(const f4*)(src + 0);
            const f4 q1 = *(const f4*)(src + 4);
            const f4 q2 = *(const f4*)(src + 8);
            const f4 q3 = *(const f4*)(src + 12);
            const v4bf c0 = __builtin_convertvector(q0, v4bf);
            const v4bf c1 = __builtin_convertvector(q1, v4bf);
            const v4bf c2 = __builtin_convertvector(q2, v4bf);
            const v4bf c3 = __builtin_convertvector(q3, v4bf);
            const v8bf h0 = __builtin_shufflevector(c0, c1, 0, 1, 2, 3, 4, 5, 6, 7);
            const v8bf h1 = __builtin_shufflevector(c2, c3, 0, 1, 2, 3, 4, 5, 6, 7);
            *(v16bf*)&bbuf[row * BK + col] = concat8(h0, h1);
        }
        __syncthreads();                     // bf16 tile visible to all waves

        const int ksbase = ks * BK;
#pragma unroll
        for (int kk = 0; kk < BK; kk += 32) {
            // A fragments: 16-bit A 16x32 layout; per lane two contiguous
            // 8-half groups at K = base + sel*8 and base + 16 + sel*8.
            v16bf afrag[4];
#pragma unroll
            for (int mt = 0; mt < 4; ++mt) {
                const int row = mgroup * 64 + mt * 16 + lrow;
                const uint16_t* ap =
                    Abf + (size_t)row * LAYER + (size_t)(ksbase + kk + sel * 8);
                afrag[mt] = concat8(*(const v8bf*)ap, *(const v8bf*)(ap + 16));
            }
#pragma unroll
            for (int nt = 0; nt < 2; ++nt) {
                // B fragment: column-major 32x16; lane holds two 8-bf16 K groups
                // at K = sel*8 and 16 + sel*8 for vocab column (ngroup*32 + nt*16 + lrow).
                const int ldsrow = ngroup * 32 + nt * 16 + lrow;
                const uint16_t* bp = &bbuf[ldsrow * BK + kk + sel * 8];
                const v16bf bfrag =
                    concat8(*(const v8bf*)bp, *(const v8bf*)(bp + 16));
#pragma unroll
                for (int mt = 0; mt < 4; ++mt) {
                    acc[mt][nt] = __builtin_amdgcn_wmma_f32_16x16x32_bf16(
                        false, afrag[mt], false, bfrag, (short)0, acc[mt][nt],
                        false, false);
                }
            }
        }
        // no trailing barrier needed: next iteration's post-wait barrier
        // orders this iteration's LDS reads against the next tile's writes.
    }

    // C layout: VGPR i -> M = mbase + sel*8 + i, N = lane&15.
#pragma unroll
    for (int mt = 0; mt < 4; ++mt) {
        const int rbase = mgroup * 64 + mt * 16 + sel * 8;
#pragma unroll
        for (int nt = 0; nt < 2; ++nt) {
            const int c = vblock + ngroup * 32 + nt * 16 + lrow;
            if (c < VOCAB) {
                const float bb = bout[c];
#pragma unroll
                for (int i = 0; i < 8; ++i)
                    logits[(size_t)(rbase + i) * VOCAB + c] = acc[mt][nt][i] + bb;
            }
        }
    }
}

// ---------------------------------------------------------------------------
// K3: per-row softmax stats (max, sum of exp). One block per row.
// ---------------------------------------------------------------------------
__global__ __launch_bounds__(256) void rowstats_kernel(
    const float* __restrict__ logits, float* __restrict__ rowmax,
    float* __restrict__ rowsum) {
    const int b = blockIdx.x;
    const int t = threadIdx.x;
    const float* row = logits + (size_t)b * VOCAB;
    __shared__ float red[256];

    float m = -3.4e38f;
    for (int i = t; i < VOCAB; i += 256) m = fmaxf(m, row[i]);
    red[t] = m;
    __syncthreads();
    for (int s = 128; s > 0; s >>= 1) {
        if (t < s) red[t] = fmaxf(red[t], red[t + s]);
        __syncthreads();
    }
    const float rm = red[0];
    __syncthreads();

    float sum = 0.f;
    for (int i = t; i < VOCAB; i += 256) sum += __expf(row[i] - rm);
    red[t] = sum;
    __syncthreads();
    for (int s = 128; s > 0; s >>= 1) {
        if (t < s) red[t] += red[t + s];
        __syncthreads();
    }
    if (t == 0) {
        rowmax[b] = rm;
        rowsum[b] = red[0];
    }
}

// ---------------------------------------------------------------------------
// K4: out = softmax(logits) * p_gen
// ---------------------------------------------------------------------------
__global__ __launch_bounds__(256) void probs_kernel(
    const float* __restrict__ logits, const float* __restrict__ rowmax,
    const float* __restrict__ rowsum, const float* __restrict__ pgen,
    float* __restrict__ out) {
    const int b = blockIdx.y;
    const int v = blockIdx.x * 256 + threadIdx.x;
    if (v >= VOCAB) return;
    const size_t idx = (size_t)b * VOCAB + v;
    out[idx] = __expf(logits[idx] - rowmax[b]) * (pgen[b] / rowsum[b]);
}

// ---------------------------------------------------------------------------
// K5: scatter-add attention copy probabilities. One block per row.
// ---------------------------------------------------------------------------
__global__ __launch_bounds__(416) void scatter_kernel(
    const float* __restrict__ att, const int* __restrict__ ids,
    const float* __restrict__ pgen, float* __restrict__ out) {
    const int b = blockIdx.x;
    const int t = threadIdx.x;
    if (t < SRC) {
        const float w = att[(size_t)b * SRC + t] * (1.f - pgen[b]);
        atomicAdd(out + (size_t)b * VOCAB + ids[(size_t)b * SRC + t], w);
    }
}

// ---------------------------------------------------------------------------
// K6: out = log(out)
// ---------------------------------------------------------------------------
__global__ __launch_bounds__(256) void log_kernel(float* __restrict__ out) {
    const int b = blockIdx.y;
    const int v = blockIdx.x * 256 + threadIdx.x;
    if (v >= VOCAB) return;
    const size_t idx = (size_t)b * VOCAB + v;
    out[idx] = __logf(out[idx]);
}

// ---------------------------------------------------------------------------
extern "C" void kernel_launch(void* const* d_in, const int* in_sizes, int n_in,
                              void* d_out, int out_size, void* d_ws,
                              size_t ws_size, hipStream_t stream) {
    const float* x    = (const float*)d_in[0];   // [256,1024]
    const float* att  = (const float*)d_in[1];   // [256,400]
    const int*   ids  = (const int*)d_in[2];     // [256,400]
    const float* Wout = (const float*)d_in[3];   // [50257,1024]
    const float* bout = (const float*)d_in[4];   // [50257]
    const float* Wgen = (const float*)d_in[5];   // [1,1024]
    const float* bgen = (const float*)d_in[6];   // [1]
    float* out = (float*)d_out;

    char* ws = (char*)d_ws;
    size_t off = 0;
    float* logits = (float*)(ws + off);
    off += (size_t)B_ROWS * VOCAB * sizeof(float);       // 51,463,168 B
    uint16_t* Abf = (uint16_t*)(ws + off);
    off += (size_t)B_ROWS * LAYER * sizeof(uint16_t);    // 524,288 B
    float* pgen = (float*)(ws + off);
    off += B_ROWS * sizeof(float);
    float* rowmax = (float*)(ws + off);
    off += B_ROWS * sizeof(float);
    float* rowsum = (float*)(ws + off);
    off += B_ROWS * sizeof(float);

    prep_kernel<<<B_ROWS, 256, 0, stream>>>(x, Wgen, bgen, Abf, pgen);

    const int nblocks = (VOCAB + BN - 1) / BN;           // 786
    gemm_kernel<<<nblocks, 256, 0, stream>>>(Abf, Wout, bout, logits);

    rowstats_kernel<<<B_ROWS, 256, 0, stream>>>(logits, rowmax, rowsum);

    dim3 eg((VOCAB + 255) / 256, B_ROWS);
    probs_kernel<<<eg, 256, 0, stream>>>(logits, rowmax, rowsum, pgen, out);

    scatter_kernel<<<B_ROWS, 416, 0, stream>>>(att, ids, pgen, out);

    log_kernel<<<eg, 256, 0, stream>>>(out);
}
